// MultiHeadAttention_90640989815096
// MI455X (gfx1250) — matrix operand order repaired and checked by reading, not compiled, not run on hardware
//
#include <hip/hip_runtime.h>

typedef __attribute__((ext_vector_type(16))) _Float16 v16h;
typedef __attribute__((ext_vector_type(8)))  float    v8f;

#define B_   2
#define T_   2048
#define D_   1024
#define H_   16
#define HD_  64
#define MTOT (B_ * T_)

static __device__ __forceinline__ v8f wmma_f16(v16h a, v16h b, v8f c) {
  // (neg_a, A, neg_b, B, c_mod, C, reuse_a, reuse_b)
  return __builtin_amdgcn_wmma_f32_16x16x32_f16(false, a, false, b, (short)0, c, false, false);
}

// ---------------------------------------------------------------------------
// Kernel 0: convert 4 weight matrices (f32, [D][D]) to transposed f16 [D][D]
// Wt[mat][d][o] = W[mat][o][d]  -> WMMA B-fragment rows become contiguous.
// ---------------------------------------------------------------------------
__global__ void wt_convert_kernel(const float* __restrict__ Wq,
                                  const float* __restrict__ Wk,
                                  const float* __restrict__ Wv,
                                  const float* __restrict__ Wo,
                                  _Float16* __restrict__ Wt) {
  unsigned idx = blockIdx.x * blockDim.x + threadIdx.x;   // 0 .. 4*2^20-1
  unsigned mat = idx >> 20;
  unsigned rem = idx & ((1u << 20) - 1u);
  unsigned d = rem >> 10;
  unsigned o = rem & 1023u;
  const float* W = (mat == 0) ? Wq : (mat == 1) ? Wk : (mat == 2) ? Wv : Wo;
  Wt[idx] = (_Float16)W[(size_t)o * D_ + d];
}

// ---------------------------------------------------------------------------
// Projection GEMM: Y = scale * (X(f32)[M x D] @ Wt(f16, pre-transposed)[D x D])
// One 16x32 tile per wave (A fragment reused by two WMMAs), 8 waves/block.
// transpose_out == 0: Y(f16)[M x D] row-major
// transpose_out == 1: Y(f16) stored as [b][feature][t]  (for K -> KhT)
// ---------------------------------------------------------------------------
__global__ void gemm_xw_f32a_kernel(const float* __restrict__ X,
                                    const _Float16* __restrict__ Wt,
                                    _Float16* __restrict__ Y, float scale,
                                    int transpose_out) {
  const int lane  = threadIdx.x & 31;
  const int wave  = threadIdx.x >> 5;
  const int tile  = blockIdx.x * 8 + wave;   // 8192 tiles: 256 m x 32 n-pairs
  const int mb    = (tile >> 5) * 16;
  const int nb    = (tile & 31) * 32;
  const int khalf = lane >> 4;
  const float* xrow = X + (size_t)(mb + (lane & 15)) * D_;

  v8f acc0 = {}, acc1 = {};
  for (int k0 = 0; k0 < D_; k0 += 32) {
    v16h a;
#pragma unroll
    for (int j = 0; j < 8; ++j) {
      a[j]     = (_Float16)xrow[k0 + khalf * 8 + j];
      a[j + 8] = (_Float16)xrow[k0 + 16 + khalf * 8 + j];
    }
    const _Float16* wcol = Wt + (size_t)k0 * D_ + nb + (lane & 15);
    v16h b0, b1;
#pragma unroll
    for (int j = 0; j < 8; ++j) {
      b0[j]     = wcol[(size_t)(khalf * 8 + j) * D_];
      b0[j + 8] = wcol[(size_t)(16 + khalf * 8 + j) * D_];
      b1[j]     = wcol[(size_t)(khalf * 8 + j) * D_ + 16];
      b1[j + 8] = wcol[(size_t)(16 + khalf * 8 + j) * D_ + 16];
    }
    if (k0 + 32 < D_)   // pull next weight tile toward the WGP (global_prefetch)
      __builtin_prefetch(Wt + (size_t)(k0 + 32 + lane) * D_ + nb, 0, 3);
    acc0 = wmma_f16(a, b0, acc0);
    acc1 = wmma_f16(a, b1, acc1);
  }
  const int m0 = mb + 8 * khalf;
  if (transpose_out) {
#pragma unroll
    for (int r = 0; r < 8; ++r) {
      const int mg = m0 + r;               // global row = b*T + t
      const int bb = mg >> 11;             // T_ == 2048
      const int tq = mg & (T_ - 1);
      const int nc0 = nb + (lane & 15);
      Y[((size_t)(bb * D_ + nc0)) * T_ + tq]      = (_Float16)(acc0[r] * scale);
      Y[((size_t)(bb * D_ + nc0 + 16)) * T_ + tq] = (_Float16)(acc1[r] * scale);
    }
  } else {
#pragma unroll
    for (int r = 0; r < 8; ++r) {
      _Float16* yrow = Y + (size_t)(m0 + r) * D_ + nb + (lane & 15);
      yrow[0]  = (_Float16)(acc0[r] * scale);
      yrow[16] = (_Float16)(acc1[r] * scale);
    }
  }
}

// ---------------------------------------------------------------------------
// Fused attention: one block = (b, h, 16-query strip).
//  stage 0: async-DMA the key mask row (8 KB) into LDS (ASYNCcnt path)
//  stage 1: scores (2 WMMA per k-tile) + bias + mask -> LDS (16 x 2048 f32)
//  stage 2: row softmax in LDS; write final attn tensor once to HBM
//  stage 3: attn @ V from LDS (K-split across wave pairs, LDS combine)
// LDS: 128 KB scores + 1 KB reduce + 8 KB partials + 8 KB mask = 145 KB
// ---------------------------------------------------------------------------
__global__ void attn_fused_kernel(const _Float16* __restrict__ Qh,
                                  const _Float16* __restrict__ KhT,
                                  const _Float16* __restrict__ Vh,
                                  const float* __restrict__ bias,
                                  const int* __restrict__ mask,
                                  const int* __restrict__ causal_p,
                                  float* __restrict__ attn,
                                  _Float16* __restrict__ OutH) {
  extern __shared__ char smem[];
  float* S     = (float*)smem;                           // [16][T_]
  float* red   = (float*)(smem + 16 * T_ * 4);           // [16][16]
  float* avp   = (float*)(smem + 16 * T_ * 4 + 1024);    // [8][256]
  int*   maskS = (int*)  (smem + 16 * T_ * 4 + 1024 + 8 * 256 * 4);  // [T_]

  const int t    = threadIdx.x;
  const int lane = t & 31;
  const int wave = t >> 5;
  int blk = blockIdx.x;                 // b*H*(T/16) + h*(T/16) + qt
  const int qt = blk & 127; blk >>= 7;
  const int h  = blk & 15;  blk >>= 4;
  const int b  = blk;
  const int qb = qt * 16;
  const int causal = *causal_p;
  const int khalf  = lane >> 4;
  const int l16    = lane & 15;

  // ---- stage 0: async copy of mask[b, :] into LDS (2 x b128 per thread) ---
  {
#pragma unroll
    for (int i = 0; i < 2; ++i) {
      // VDST holds the LDS byte offset (low 32 bits of the generic pointer);
      // VADDR holds the 64-bit global address.
      unsigned ldsoff = (unsigned)(size_t)(maskS + i * 1024 + t * 4);
      const int* gptr = mask + b * T_ + i * 1024 + t * 4;
      asm volatile("global_load_async_to_lds_b128 %0, %1, off"
                   :: "v"(ldsoff), "v"(gptr) : "memory");
    }
    asm volatile("s_wait_asynccnt 0x0" ::: "memory");
  }
  __syncthreads();

  // ---- stage 1: scores into LDS -------------------------------------------
  // A fragments (Q strip) loaded once, reused for all 16 k-tiles of this wave.
  const _Float16* qrow = Qh + (size_t)(b * T_ + qb + l16) * D_ + h * HD_;
  v16h a0, a1;
#pragma unroll
  for (int j = 0; j < 8; ++j) {
    a0[j]     = qrow[khalf * 8 + j];
    a0[j + 8] = qrow[16 + khalf * 8 + j];
    a1[j]     = qrow[32 + khalf * 8 + j];
    a1[j + 8] = qrow[48 + khalf * 8 + j];
  }
  const _Float16* ktbase = KhT + (size_t)(b * D_ + h * HD_) * T_;
  for (int i = 0; i < 16; ++i) {
    const int kb = (wave + i * 8) * 16;
    // B[k][n] = K[b, kb+n, h*HD+k] : lane holds column n = l16, elements over k (strided rows of KhT)
    v16h b0, b1;
#pragma unroll
    for (int j = 0; j < 8; ++j) {
      b0[j]     = ktbase[(size_t)(khalf * 8 + j) * T_ + kb + l16];
      b0[j + 8] = ktbase[(size_t)(16 + khalf * 8 + j) * T_ + kb + l16];
      b1[j]     = ktbase[(size_t)(32 + khalf * 8 + j) * T_ + kb + l16];
      b1[j + 8] = ktbase[(size_t)(48 + khalf * 8 + j) * T_ + kb + l16];
    }
    v8f acc = {};
    acc = wmma_f16(a0, b0, acc);
    acc = wmma_f16(a1, b1, acc);
    const int nc = kb + l16;
    const int am = maskS[nc];
#pragma unroll
    for (int r = 0; r < 8; ++r) {
      const int qrel = 8 * khalf + r;
      const int q = qb + qrel;
      const float s = acc[r] + bias[((size_t)h * T_ + q) * T_ + nc];
      const bool ok = (am != 0) && (!causal || (nc <= q));
      S[qrel * T_ + nc] = ok ? s : -1.0e9f;
    }
  }
  __syncthreads();

  // ---- stage 2: softmax over each of the 16 rows --------------------------
  {
    const int r  = t >> 4;         // row 0..15
    const int c0 = t & 15;         // this thread handles cols c0 + 16*i
    float* Srow = S + r * T_;
    float m = -3.4e38f;
    for (int i = 0; i < 128; ++i) m = fmaxf(m, Srow[c0 + i * 16]);
    red[r * 16 + c0] = m;
    __syncthreads();
#pragma unroll
    for (int i = 0; i < 16; ++i) m = fmaxf(m, red[r * 16 + i]);
    __syncthreads();
    float sum = 0.f;
    for (int i = 0; i < 128; ++i) {
      const float e = __expf(Srow[c0 + i * 16] - m);
      Srow[c0 + i * 16] = e;
      sum += e;
    }
    red[r * 16 + c0] = sum;
    __syncthreads();
    float tot = 0.f;
#pragma unroll
    for (int i = 0; i < 16; ++i) tot += red[r * 16 + i];
    const float inv = 1.0f / tot;
    float* arow = attn + ((size_t)(b * H_ + h) * T_ + qb + r) * T_;
    for (int i = 0; i < 128; ++i) {
      const float v = Srow[c0 + i * 16] * inv;
      Srow[c0 + i * 16] = v;       // kept for stage 3
      arow[c0 + i * 16] = v;       // the attn output, written exactly once
    }
  }
  __syncthreads();

  // ---- stage 3: out_h = attn(S in LDS) @ V --------------------------------
  // wave w: d-tile (w&3), K-range half (w>>2); partials combined via LDS.
  {
    const int db = (wave & 3) * 16;
    const int k0base = (wave >> 2) * (T_ / 2);
    const float* arowS = S + l16 * T_;
    v8f acc = {};
    for (int k0 = k0base; k0 < k0base + T_ / 2; k0 += 32) {
      v16h af;
#pragma unroll
      for (int j = 0; j < 8; ++j) {
        af[j]     = (_Float16)arowS[k0 + khalf * 8 + j];
        af[j + 8] = (_Float16)arowS[k0 + 16 + khalf * 8 + j];
      }
      const _Float16* vcol = Vh + (size_t)(b * T_ + k0) * D_ + h * HD_ + db + l16;
      v16h bf;
#pragma unroll
      for (int j = 0; j < 8; ++j) {
        bf[j]     = vcol[(size_t)(khalf * 8 + j) * D_];
        bf[j + 8] = vcol[(size_t)(16 + khalf * 8 + j) * D_];
      }
      acc = wmma_f16(af, bf, acc);
    }
#pragma unroll
    for (int r = 0; r < 8; ++r)
      avp[wave * 256 + (8 * khalf + r) * 16 + l16] = acc[r];
  }
  __syncthreads();
  {
#pragma unroll
    for (int e = 0; e < 4; ++e) {
      const float v = avp[e * 256 + t] + avp[(e + 4) * 256 + t];
      const int m = t >> 4, n = t & 15;
      OutH[(size_t)(b * T_ + qb + m) * D_ + h * HD_ + e * 16 + n] = (_Float16)v;
    }
  }
}

// ---------------------------------------------------------------------------
// Final projection: out(f32) = OutH(f16) @ Wo^T(f16), 16x32 tile per wave
// ---------------------------------------------------------------------------
__global__ void gemm_ow_kernel(const _Float16* __restrict__ X,
                               const _Float16* __restrict__ Wt,
                               float* __restrict__ Y) {
  const int lane  = threadIdx.x & 31;
  const int wave  = threadIdx.x >> 5;
  const int tile  = blockIdx.x * 8 + wave;
  const int mb    = (tile >> 5) * 16;
  const int nb    = (tile & 31) * 32;
  const int khalf = lane >> 4;
  const _Float16* xrow = X + (size_t)(mb + (lane & 15)) * D_;

  v8f acc0 = {}, acc1 = {};
  for (int k0 = 0; k0 < D_; k0 += 32) {
    v16h a;
#pragma unroll
    for (int j = 0; j < 8; ++j) {
      a[j]     = xrow[k0 + khalf * 8 + j];
      a[j + 8] = xrow[k0 + 16 + khalf * 8 + j];
    }
    const _Float16* wcol = Wt + (size_t)k0 * D_ + nb + (lane & 15);
    v16h b0, b1;
#pragma unroll
    for (int j = 0; j < 8; ++j) {
      b0[j]     = wcol[(size_t)(khalf * 8 + j) * D_];
      b0[j + 8] = wcol[(size_t)(16 + khalf * 8 + j) * D_];
      b1[j]     = wcol[(size_t)(khalf * 8 + j) * D_ + 16];
      b1[j + 8] = wcol[(size_t)(16 + khalf * 8 + j) * D_ + 16];
    }
    if (k0 + 32 < D_)
      __builtin_prefetch(Wt + (size_t)(k0 + 32 + lane) * D_ + nb, 0, 3);
    acc0 = wmma_f16(a, b0, acc0);
    acc1 = wmma_f16(a, b1, acc1);
  }
  const int m0 = mb + 8 * khalf;
#pragma unroll
  for (int r = 0; r < 8; ++r) {
    float* yrow = Y + (size_t)(m0 + r) * D_ + nb + (lane & 15);
    yrow[0]  = acc0[r];
    yrow[16] = acc1[r];
  }
}

// ---------------------------------------------------------------------------
extern "C" void kernel_launch(void* const* d_in, const int* in_sizes, int n_in,
                              void* d_out, int out_size, void* d_ws, size_t ws_size,
                              hipStream_t stream) {
  (void)in_sizes; (void)n_in; (void)out_size; (void)ws_size;

  const float* x_q   = (const float*)d_in[0];
  const float* x_k   = (const float*)d_in[1];
  const float* x_v   = (const float*)d_in[2];
  const int*   mask  = (const int*)d_in[3];
  const int*   caus  = (const int*)d_in[4];
  const float* bias  = (const float*)d_in[5];
  const float* Wq    = (const float*)d_in[6];
  const float* Wk    = (const float*)d_in[7];
  const float* Wv    = (const float*)d_in[8];
  const float* Wo    = (const float*)d_in[9];

  char* ws = (char*)d_ws;
  _Float16* Wt   = (_Float16*)ws;                       // 4 x 2 MiB transposed weights
  _Float16* Qh   = (_Float16*)(ws + (8u  << 20));       // 8 MiB, [b*T+t][feat], scaled
  _Float16* KhT  = (_Float16*)(ws + (16u << 20));       // 8 MiB, [b][feat][t]
  _Float16* Vh   = (_Float16*)(ws + (24u << 20));       // 8 MiB, [b*T+t][feat]
  _Float16* OutH = (_Float16*)(ws + (32u << 20));       // 8 MiB

  float* out  = (float*)d_out;
  float* attn = out + (size_t)MTOT * D_;                // (B,H,T,T) region of d_out

  // 0) weight transpose+convert
  wt_convert_kernel<<<(4u << 20) / 256, 256, 0, stream>>>(Wq, Wk, Wv, Wo, Wt);

  // 1) Q,K,V projections. Softmax 1/sqrt(HD) folded into Q; K stored transposed.
  gemm_xw_f32a_kernel<<<1024, 256, 0, stream>>>(x_q, Wt,              Qh,  0.125f, 0);
  gemm_xw_f32a_kernel<<<1024, 256, 0, stream>>>(x_k, Wt + (1u << 20), KhT, 1.0f,   1);
  gemm_xw_f32a_kernel<<<1024, 256, 0, stream>>>(x_v, Wt + (2u << 20), Vh,  1.0f,   0);

  // 2) fused scores + softmax + AV: one block per (b, h, 16-query strip)
  const size_t lds_bytes = 16 * T_ * 4 + 1024 + 8 * 256 * 4 + T_ * 4;  // 148480
  attn_fused_kernel<<<B_ * H_ * (T_ / 16), 256, lds_bytes, stream>>>(
      Qh, KhT, Vh, bias, mask, caus, attn, OutH);

  // 3) output projection -> f32 out
  gemm_ow_kernel<<<1024, 256, 0, stream>>>(OutH, Wt + (3u << 20), out);
}
